// LSTM_41815801594203
// MI455X (gfx1250) — matrix operand order repaired and checked by reading, not compiled, not run on hardware
//
#include <hip/hip_runtime.h>
#include <hip/hip_bf16.h>

// ---------------- CDNA5 WMMA types ----------------
typedef __bf16 bf16_t;
typedef bf16_t v16bf __attribute__((ext_vector_type(16)));
typedef bf16_t v8bf  __attribute__((ext_vector_type(8)));
typedef float  v8f   __attribute__((ext_vector_type(8)));

#define B_DIM 64
#define T_DIM 512
#define I_DIM 512
#define H_DIM 1024
#define O_DIM 256
#define GH    (4 * H_DIM)          // 4096 gate columns

#define NWG   64                   // workgroups in recurrent kernel (H/16)
#define LWS   1032                 // padded LDS row stride (bf16 elems)
#define GS    68                   // padded LDS row stride (f32) for gates scratch

static __device__ __forceinline__ float sigmoidf_(float x) {
    return 1.0f / (1.0f + __expf(-x));
}

static __device__ __forceinline__ v8f vzero8() {
    v8f z;
#pragma unroll
    for (int i = 0; i < 8; ++i) z[i] = 0.0f;
    return z;
}

// CDNA5 async copy: 16 bytes per lane, global -> LDS, tracked by ASYNCcnt.
static __device__ __forceinline__ void async_copy_b128(void* lds_dst, const void* gsrc) {
    unsigned lds = (unsigned)(uintptr_t)lds_dst;   // low 32 bits = LDS offset
    asm volatile("global_load_async_to_lds_b128 %0, %1, off"
                 :: "v"(lds), "v"(gsrc) : "memory");
}
static __device__ __forceinline__ void wait_asynccnt0() {
    asm volatile("s_wait_asynccnt 0x0" ::: "memory");
}

// A fragment (16x32 bf16, M x K): lane m=lane&15, half=lane>>4.
// VGPR j<4 holds K=2j,2j+1 (+8*half); VGPR j>=4 holds K=16+2(j-4),+1 (+8*half)
// => two contiguous 16B loads at k0+8*half and k0+16+8*half.
static __device__ __forceinline__ v16bf load_a_frag(const bf16_t* __restrict__ A,
                                                    int row_stride, int m_base,
                                                    int k0, int lane) {
    const int m    = m_base + (lane & 15);
    const int half = lane >> 4;
    const bf16_t* p = A + (size_t)m * row_stride + k0 + 8 * half;
    v16bf a;
    *(v8bf*)&a       = *(const v8bf*)p;
    *((v8bf*)&a + 1) = *(const v8bf*)(p + 16);
    return a;
}

// B fragment (32x16 bf16, K x N) from row-major weights W[n][k]:
// lane n=lane&15, koff=k0+8*(lane>>4); same K order as the A fragment
// => two contiguous 16B loads at koff and koff+16 along k of weight row n.
static __device__ __forceinline__ v16bf load_b_frag(const bf16_t* __restrict__ W,
                                                    int row_stride, int n_base,
                                                    int k0, int lane) {
    const int n    = n_base + (lane & 15);
    const int koff = k0 + 8 * (lane >> 4);
    const bf16_t* p = W + (size_t)n * row_stride + koff;
    v16bf b;
    *(v8bf*)&b       = *(const v8bf*)p;
    *((v8bf*)&b + 1) = *(const v8bf*)(p + 16);
    return b;
}

static __device__ __forceinline__ v8f wmma_bf16(v16bf a, v16bf b, v8f c) {
    return __builtin_amdgcn_wmma_f32_16x16x32_bf16(false, a, false, b,
                                                   (short)0, c, false, false);
}

// ---------------- conversion kernels ----------------
__global__ void convert_x_kernel(const float* __restrict__ x, bf16_t* __restrict__ xb) {
    size_t i = (size_t)blockIdx.x * blockDim.x + threadIdx.x;
    const size_t n = (size_t)B_DIM * T_DIM * I_DIM;
    if (i >= n) return;
    int    ii = (int)(i % I_DIM);
    size_t bt = i / I_DIM;
    int    t  = (int)(bt % T_DIM);
    int    b  = (int)(bt / T_DIM);
    // xb[(t*B + b)][i] = x[b][t][i]  (rows ordered for the xg GEMM)
    xb[((size_t)t * B_DIM + b) * I_DIM + ii] = (bf16_t)x[i];
}

__global__ void f32_to_bf16_kernel(const float* __restrict__ src,
                                   bf16_t* __restrict__ dst, size_t n) {
    size_t i = (size_t)blockIdx.x * blockDim.x + threadIdx.x;
    if (i < n) dst[i] = (bf16_t)src[i];
}

// ---------------- generic WMMA GEMM: C[M][N] = A[M][K] * W[N][K]^T + bias ----------------
// Block tile 64(M) x 128(N), 8 waves; wave w: M-tile = w&3, N-tiles = (w>>2)*4 .. +3.
__global__ __launch_bounds__(256) void gemm_bias_kernel(
    const bf16_t* __restrict__ A, const bf16_t* __restrict__ W,
    const float* __restrict__ bias0, const float* __restrict__ bias1,
    float* __restrict__ C, int M, int N, int K) {
    const int lane = threadIdx.x & 31;
    const int wave = threadIdx.x >> 5;
    const int m0 = blockIdx.y * 64 + (wave & 3) * 16;
    const int n0 = blockIdx.x * 128 + (wave >> 2) * 64;
    (void)M;

    v8f acc[4];
#pragma unroll
    for (int t = 0; t < 4; ++t) acc[t] = vzero8();

    for (int kb = 0; kb < K; kb += 32) {
        v16bf a = load_a_frag(A, K, m0, kb, lane);
#pragma unroll
        for (int tn = 0; tn < 4; ++tn) {
            v16bf b = load_b_frag(W, K, n0 + tn * 16, kb, lane);
            acc[tn] = wmma_bf16(a, b, acc[tn]);
        }
    }

    const int half = lane >> 4;
    const int nl   = lane & 15;
#pragma unroll
    for (int tn = 0; tn < 4; ++tn) {
        const int col = n0 + tn * 16 + nl;
        float bs = bias0 ? bias0[col] : 0.0f;
        if (bias1) bs += bias1[col];
#pragma unroll
        for (int r = 0; r < 8; ++r) {
            const int row = m0 + r + 8 * half;
            C[(size_t)row * N + col] = acc[tn][r] + bs;
        }
    }
}

// ---------------- grid-wide generation barrier ----------------
static __device__ __forceinline__ void gridbar(unsigned* cnt, unsigned* gen,
                                               unsigned target, unsigned nwg) {
    __syncthreads();
    if (threadIdx.x == 0) {
        __threadfence();
        unsigned prev = __hip_atomic_fetch_add(cnt, 1u, __ATOMIC_ACQ_REL,
                                               __HIP_MEMORY_SCOPE_AGENT);
        if (prev == nwg - 1) {
            __hip_atomic_store(cnt, 0u, __ATOMIC_RELAXED, __HIP_MEMORY_SCOPE_AGENT);
            __hip_atomic_fetch_add(gen, 1u, __ATOMIC_RELEASE,
                                   __HIP_MEMORY_SCOPE_AGENT);
        }
        while (__hip_atomic_load(gen, __ATOMIC_ACQUIRE,
                                 __HIP_MEMORY_SCOPE_AGENT) < target) {
            __builtin_amdgcn_s_sleep(2);
        }
    }
    __syncthreads();
}

// ---------------- persistent recurrent LSTM kernel ----------------
// 64 WGs x 256 threads. WG g owns h columns [g*16, g*16+16).
// LDS: W_hh slice (64 rows, bf16) + full h tile (64 rows, bf16, async-staged
// per step) + gates scratch + c slice  ~= 279 KB of the 320 KB WGP LDS.
__global__ __launch_bounds__(256) void lstm_recurrent_kernel(
    const float* __restrict__ xg,    // [T][B][4H] f32, biases included
    const bf16_t* __restrict__ whh,  // [4H][H] bf16
    bf16_t* __restrict__ hbuf,       // [2][B][H] bf16 ping-pong
    unsigned* __restrict__ bar_cnt, unsigned* __restrict__ bar_gen, int T) {
    extern __shared__ char smem[];
    bf16_t* lw  = (bf16_t*)smem;                 // [64][LWS] weights
    bf16_t* lh  = lw + (size_t)64 * LWS;         // [64][LWS] h (A matrix)
    float*  gsm = (float*)(lh + (size_t)64 * LWS);  // [64][GS]
    float*  csm = gsm + 64 * GS;                 // [64][16]

    const int tid  = threadIdx.x;
    const int lane = tid & 31;
    const int wave = tid >> 5;
    const int g    = blockIdx.x;

    // Stage own W_hh slice into LDS (bf16, 128 KB), reused for all T steps.
    // n_local = gate*16 + q  ->  global row gate*H + g*16 + q
    for (int idx = tid; idx < 64 * (H_DIM / 8); idx += 256) {
        const int row  = idx >> 7;   // 0..63
        const int ch   = idx & 127;  // 128 chunks of 8 bf16
        const int gate = row >> 4, q = row & 15;
        async_copy_b128(lw + (size_t)row * LWS + ch * 8,
                        whh + (size_t)(gate * H_DIM + g * 16 + q) * H_DIM + ch * 8);
    }
    for (int idx = tid; idx < 64 * 16; idx += 256) csm[idx] = 0.0f;
    // zero own slice of h buffer 0
    for (int idx = tid; idx < 64 * 16; idx += 256) {
        const int b = idx >> 4, q = idx & 15;
        hbuf[(size_t)b * H_DIM + g * 16 + q] = (bf16_t)0.0f;
    }
    wait_asynccnt0();
    gridbar(bar_cnt, bar_gen, 1u, NWG);

    const int mt   = wave & 3;          // M tile (batch rows 16*mt..)
    const int nt0  = (wave >> 2) * 2;   // this wave's two N tiles
    const int half = lane >> 4;
    const int nl   = lane & 15;

    for (int t = 0; t < T; ++t) {
        const bf16_t* cur = hbuf + (size_t)(t & 1) * B_DIM * H_DIM;
        bf16_t*       nxt = hbuf + (size_t)((t + 1) & 1) * B_DIM * H_DIM;

        // async-stage full h (64 x 1024 bf16 = 128 KB) into LDS
        for (int idx = tid; idx < 64 * (H_DIM / 8); idx += 256) {
            const int b  = idx >> 7;   // batch row (wave-uniform: good coalescing)
            const int ch = idx & 127;
            async_copy_b128(lh + (size_t)b * LWS + ch * 8,
                            cur + (size_t)b * H_DIM + ch * 8);
        }
        wait_asynccnt0();
        __syncthreads();

        v8f acc0 = vzero8();
        v8f acc1 = vzero8();
        for (int kb = 0; kb < H_DIM; kb += 32) {
            v16bf a  = load_a_frag(lh, LWS, mt * 16, kb, lane);      // h from LDS
            v16bf b0 = load_b_frag(lw, LWS, nt0 * 16, kb, lane);     // W from LDS
            v16bf b1 = load_b_frag(lw, LWS, (nt0 + 1) * 16, kb, lane);
            acc0 = wmma_bf16(a, b0, acc0);
            acc1 = wmma_bf16(a, b1, acc1);
        }
#pragma unroll
        for (int r = 0; r < 8; ++r) {
            const int row = mt * 16 + r + 8 * half;
            gsm[row * GS + nt0 * 16 + nl]       = acc0[r];
            gsm[row * GS + (nt0 + 1) * 16 + nl] = acc1[r];
        }
        __syncthreads();

        // gate math on own slice: 64 b x 16 q items
        const float* xgt = xg + (size_t)t * B_DIM * GH;
        for (int idx = tid; idx < 1024; idx += 256) {
            const int b = idx >> 4, q = idx & 15;
            const int col = g * 16 + q;
            const float ig = gsm[b * GS + q]      + xgt[(size_t)b * GH + col];
            const float fg = gsm[b * GS + 16 + q] + xgt[(size_t)b * GH + H_DIM + col];
            const float gg = gsm[b * GS + 32 + q] + xgt[(size_t)b * GH + 2 * H_DIM + col];
            const float og = gsm[b * GS + 48 + q] + xgt[(size_t)b * GH + 3 * H_DIM + col];
            const float c  = csm[idx];
            const float cn = sigmoidf_(fg) * c + sigmoidf_(ig) * tanhf(gg);
            csm[idx] = cn;
            const float h = sigmoidf_(og) * tanhf(cn);
            nxt[(size_t)b * H_DIM + col] = (bf16_t)h;
            if (t + 1 < T) {  // warm next step's preactivations (global_prefetch_b8)
                __builtin_prefetch(xgt + (size_t)B_DIM * GH + (size_t)b * GH + col, 0, 3);
            }
        }
        gridbar(bar_cnt, bar_gen, (unsigned)(t + 2), NWG);
    }
}

// ---------------- host launcher ----------------
extern "C" void kernel_launch(void* const* d_in, const int* in_sizes, int n_in,
                              void* d_out, int out_size, void* d_ws, size_t ws_size,
                              hipStream_t stream) {
    (void)in_sizes; (void)n_in; (void)out_size; (void)ws_size;
    const float* x     = (const float*)d_in[0];
    const float* W_ih  = (const float*)d_in[1];
    const float* W_hh  = (const float*)d_in[2];
    const float* b_ih  = (const float*)d_in[3];
    const float* b_hh  = (const float*)d_in[4];
    const float* W_out = (const float*)d_in[5];
    const float* b_out = (const float*)d_in[6];
    float* out = (float*)d_out;

    char* ws = (char*)d_ws;
    size_t off = 0;
    bf16_t* xb     = (bf16_t*)(ws + off); off += (size_t)B_DIM * T_DIM * I_DIM * 2;
    bf16_t* wih_b  = (bf16_t*)(ws + off); off += (size_t)GH * I_DIM * 2;
    bf16_t* whh_b  = (bf16_t*)(ws + off); off += (size_t)GH * H_DIM * 2;
    bf16_t* wout_b = (bf16_t*)(ws + off); off += (size_t)O_DIM * H_DIM * 2;
    float*  xg     = (float*)(ws + off);  off += (size_t)T_DIM * B_DIM * GH * 4;
    bf16_t* hbuf   = (bf16_t*)(ws + off); off += (size_t)2 * B_DIM * H_DIM * 2;
    unsigned* bar  = (unsigned*)(ws + off); off += 256;

    hipMemsetAsync(bar, 0, 256, stream);

    {   // x -> bf16 with [t][b] row order
        size_t n = (size_t)B_DIM * T_DIM * I_DIM;
        convert_x_kernel<<<(unsigned)((n + 255) / 256), 256, 0, stream>>>(x, xb);
    }
    {   size_t n = (size_t)GH * I_DIM;
        f32_to_bf16_kernel<<<(unsigned)((n + 255) / 256), 256, 0, stream>>>(W_ih, wih_b, n);
    }
    {   size_t n = (size_t)GH * H_DIM;
        f32_to_bf16_kernel<<<(unsigned)((n + 255) / 256), 256, 0, stream>>>(W_hh, whh_b, n);
    }
    {   size_t n = (size_t)O_DIM * H_DIM;
        f32_to_bf16_kernel<<<(unsigned)((n + 255) / 256), 256, 0, stream>>>(W_out, wout_b, n);
    }

    // xg[T*B][4H] = xb[T*B][I] @ W_ih^T + (b_ih + b_hh)
    {   dim3 grid(GH / 128, (T_DIM * B_DIM) / 64);
        gemm_bias_kernel<<<grid, 256, 0, stream>>>(xb, wih_b, b_ih, b_hh, xg,
                                                   T_DIM * B_DIM, GH, I_DIM);
    }

    // persistent recurrent scan
    {   size_t smem = (size_t)64 * LWS * 2 * 2   // weights + h tile
                    + (size_t)64 * GS * 4        // gates
                    + (size_t)64 * 16 * 4;       // c
        lstm_recurrent_kernel<<<NWG, 256, smem, stream>>>(xg, whh_b, hbuf,
                                                          bar, bar + 1, T_DIM);
    }

    // out[B][O] = h_last @ W_out^T + b_out  (h_last lands in hbuf buffer 0: T even)
    {   dim3 grid(O_DIM / 128, B_DIM / 64);
        gemm_bias_kernel<<<grid, 256, 0, stream>>>(hbuf, wout_b, b_out, nullptr,
                                                   out, B_DIM, O_DIM, H_DIM);
    }
}
